// QEffPrefillOnlyDeepseekV3MoE_23871428231439
// MI455X (gfx1250) — compile-verified, matrix operand order checked by reading, not run on hardware
//
#include <hip/hip_runtime.h>
#include <math.h>

// ---------------------------------------------------------------------------
// DeepseekV3 MoE prefill for MI455X (gfx1250, wave32, WMMA + TDM).
// Sparse routed experts (top-6 of 64) + shared SwiGLU expert.
// bf16 WMMA (v_wmma_f32_16x16x32_bf16) with fp32 accumulate.
// Weight tiles streamed global->LDS by the Tensor Data Mover (double-buffered),
// synchronized with s_wait_tensorcnt + workgroup barriers.
// ---------------------------------------------------------------------------

typedef __bf16 bf16;
typedef bf16  v16bf __attribute__((ext_vector_type(16)));
typedef float v8f   __attribute__((ext_vector_type(8)));
typedef unsigned int u32x4 __attribute__((ext_vector_type(4)));
typedef int          i32x4 __attribute__((ext_vector_type(4)));
typedef int          i32x8 __attribute__((ext_vector_type(8)));

#define T_TOK   4096
#define DDIM    1024
#define NEXP    64
#define TOPK    6
#define HDIM    768
#define SHDIM   1536
#define PMAX    32768          // padded pair capacity (24576 + 64*127 <= 32704)
#define MTILES  256            // PMAX / 128
#define ROUTED_SCALE 2.5f

// ------------------------- Tensor Data Mover helper -------------------------
// 2D tile load: rows x cols fp32 tile from a row-major matrix into LDS.
// D# built per CDNA5 ISA 8.3/8.4: group0 = {count|flags, lds_addr,
// global_addr[31:0], global_addr[56:32]|type=2}; group1 packs data_size,
// tensor_dim0/1, tile_dim0/1, tensor_dim0_stride. Groups 2/3 (3D-5D dims) and
// the trailing descriptor words are zero for a 2D tile.

__device__ __forceinline__ void tdm_load_tile_f32(unsigned lds_off,
                                                  const float* gptr,
                                                  unsigned row_elems,  // tensor_dim0 == dim0 stride
                                                  unsigned rows,       // tensor_dim1
                                                  unsigned tile_n,     // tile_dim0 (contiguous)
                                                  unsigned tile_k) {   // tile_dim1
    const unsigned long long ga = (unsigned long long)(uintptr_t)gptr;
    u32x4 g0;
    g0[0] = 1u;                                                    // count=1, user mode
    g0[1] = lds_off;                                               // D#.lds_addr
    g0[2] = (unsigned)ga;                                          // global_addr[31:0]
    g0[3] = (unsigned)((ga >> 32) & 0x01ffffffu) | (2u << 30);     // [56:32] | type=2
    i32x8 g1;
    g1[0] = (int)(2u << 16);                                       // data_size=2 (4 bytes)
    g1[1] = (int)((row_elems & 0xffffu) << 16);                    // tensor_dim0[15:0]
    g1[2] = (int)((row_elems >> 16) | ((rows & 0xffffu) << 16));   // dim0 hi | dim1 lo
    g1[3] = (int)((rows >> 16) | (tile_n << 16));                  // dim1 hi | tile_dim0
    g1[4] = (int)tile_k;                                           // tile_dim1 (tile_dim2=0)
    g1[5] = (int)row_elems;                                        // tensor_dim0_stride[31:0]
    g1[6] = 0;                                                     // stride0 hi | stride1 lo
    g1[7] = 0;
    const i32x4 z4 = {0, 0, 0, 0};
    const i32x8 z8 = {0, 0, 0, 0, 0, 0, 0, 0};
    __builtin_amdgcn_tensor_load_to_lds(g0, g1, z4, z4, z8, 0);
}

// ------------------------------- gating ------------------------------------

__global__ __launch_bounds__(64)
void init_kernel(int* expert_count) {
    expert_count[threadIdx.x] = 0;
}

__global__ __launch_bounds__(64)
void gate_topk_kernel(const float* __restrict__ x,
                      const float* __restrict__ gw,
                      const float* __restrict__ gb,
                      int* __restrict__ topk_idx,
                      float* __restrict__ topk_w,
                      int* __restrict__ expert_count) {
    __shared__ float xs[DDIM];
    __shared__ float sc[NEXP];
    __shared__ float sb[NEXP];
    const int t = blockIdx.x;
    for (int i = threadIdx.x; i < DDIM; i += 64) xs[i] = x[(size_t)t * DDIM + i];
    __syncthreads();
    const int e = threadIdx.x;
    float acc = 0.f;
    #pragma unroll 8
    for (int d = 0; d < DDIM; ++d) acc = fmaf(xs[d], gw[(size_t)d * NEXP + e], acc);
    const float s = __builtin_amdgcn_rcpf(1.f + __expf(-acc));   // sigmoid score
    sc[e] = s;
    sb[e] = s + gb[e];                                 // selection score (bias-corrected)
    __syncthreads();
    if (threadIdx.x == 0) {
        int   idx[TOPK];
        float w[TOPK];
        float sum = 0.f;
        for (int k = 0; k < TOPK; ++k) {
            int best = 0; float bv = -1e30f;
            for (int j = 0; j < NEXP; ++j)
                if (sb[j] > bv) { bv = sb[j]; best = j; }
            idx[k] = best; w[k] = sc[best]; sum += sc[best];
            sb[best] = -1e30f;
        }
        const float inv = ROUTED_SCALE / (sum + 1e-20f);
        for (int k = 0; k < TOPK; ++k) {
            topk_idx[t * TOPK + k] = idx[k];
            topk_w[t * TOPK + k]   = w[k] * inv;
            atomicAdd(&expert_count[idx[k]], 1);
        }
    }
}

__global__ __launch_bounds__(256)
void scan_kernel(const int* __restrict__ expert_count,
                 int* __restrict__ pad_off,
                 int* __restrict__ cursor,
                 int* __restrict__ tile_expert,
                 int* __restrict__ pair_token,
                 float* __restrict__ pair_weight) {
    if (threadIdx.x == 0) {
        int off = 0;
        for (int e = 0; e < NEXP; ++e) {
            pad_off[e] = off;
            cursor[e]  = off;
            off += (expert_count[e] + 127) & ~127;     // 128-align each expert bin
        }
        pad_off[NEXP] = off;
        int tile = 0;
        for (int e = 0; e < NEXP; ++e) {
            const int nt = (pad_off[e + 1] - pad_off[e]) >> 7;
            for (int i = 0; i < nt; ++i) tile_expert[tile++] = e;
        }
        for (; tile < MTILES; ++tile) tile_expert[tile] = -1;
    }
    __syncthreads();
    for (int i = threadIdx.x; i < PMAX; i += 256) {    // padding slots: token 0, weight 0
        pair_token[i]  = 0;
        pair_weight[i] = 0.f;
    }
}

__global__ __launch_bounds__(256)
void scatter_kernel(const int* __restrict__ topk_idx,
                    const float* __restrict__ topk_w,
                    int* __restrict__ cursor,
                    int* __restrict__ pair_token,
                    float* __restrict__ pair_weight) {
    const int i = blockIdx.x * 256 + threadIdx.x;      // i < T_TOK*TOPK = 24576
    const int e = topk_idx[i];
    const int p = atomicAdd(&cursor[e], 1);
    pair_token[p]  = i / TOPK;
    pair_weight[p] = topk_w[i];
}

// ------------------------- fp32 -> bf16 activation --------------------------

__global__ __launch_bounds__(256)
void xcvt_kernel(const float* __restrict__ x, bf16* __restrict__ xb) {
    const size_t i = ((size_t)blockIdx.x * 256 + threadIdx.x) * 4;
    const float4 v = *(const float4*)(x + i);
    union { bf16 b[4]; uint2 u; } o;
    o.b[0] = (bf16)v.x; o.b[1] = (bf16)v.y; o.b[2] = (bf16)v.z; o.b[3] = (bf16)v.w;
    *(uint2*)(xb + i) = o.u;
}

// ------------------------- WMMA fragment helpers ----------------------------
// LDS tiles are [rows][32] bf16, row-major. Per the CDNA5 ISA 16-bit A/B
// layout, lane (r = lane&15, half = lane>>4) holds two contiguous 8-element
// K runs: [half*8, half*8+8) and [16+half*8, ...): two ds_load_b128 per frag.

__device__ __forceinline__ v16bf frag_from_lds(const bf16* base, int row0, int lane) {
    const int half = (lane >> 4) & 1;
    const int r    = lane & 15;
    const bf16* p  = base + (size_t)(row0 + r) * 32 + half * 8;
    union { v16bf v; uint4 u[2]; } f;
    f.u[0] = *(const uint4*)(p);
    f.u[1] = *(const uint4*)(p + 16);
    return f.v;
}

#define WMMA_BF16(A, B, C) \
    __builtin_amdgcn_wmma_f32_16x16x32_bf16(false, (A), false, (B), (short)0, (C), false, false)

__device__ __forceinline__ float silu_fast(float g) {
    return g * __builtin_amdgcn_rcpf(1.f + __expf(-g));
}

// ------------------------- gate/up GEMM + SwiGLU ----------------------------
// C tile: 128 rows x 64 cols, 8 waves (4x2 of 32x32), K = 1024 in 32-chunks.
// Weight tiles (fp32) streamed into LDS by the TDM, double-buffered; conversion
// to bf16 (transposed [n][k]) happens LDS->LDS so WMMA frags stay ds_load_b128.
// GATHER=true: rows are routed pairs (token gathered, expert per M-tile).

template <bool GATHER>
__global__ __launch_bounds__(256)
void gateup_kernel(const bf16* __restrict__ X,
                   const float* __restrict__ Gall,
                   const float* __restrict__ Uall,
                   const int* __restrict__ pair_token,
                   const int* __restrict__ tile_expert,
                   bf16* __restrict__ Hout,
                   int Ntot) {
    __shared__ __align__(16) float stG[2][32 * 64];   // TDM landing zone (fp32)
    __shared__ __align__(16) float stU[2][32 * 64];
    __shared__ __align__(16) bf16  lA [128 * 32];
    __shared__ __align__(16) bf16  lBg[64 * 32];      // transposed: [n][k]
    __shared__ __align__(16) bf16  lBu[64 * 32];

    const int bn = blockIdx.x;
    const int bm = blockIdx.y;
    const int n0 = bn * 64;

    const float* G;
    const float* U;
    if (GATHER) {
        const int e = tile_expert[bm];
        if (e < 0) return;                             // whole block exits together
        const size_t off = (size_t)e * DDIM * Ntot;
        G = Gall + off; U = Uall + off;
    } else {
        G = Gall; U = Uall;
    }

    const int tid  = threadIdx.x;
    const int lane = tid & 31;
    const int wave = tid >> 5;
    const int wm   = wave >> 1;                        // 0..3
    const int wn   = wave & 1;                         // 0..1

    const int arow  = tid >> 1;                        // 0..127
    const int ahalf = tid & 1;
    int tok;
    if (GATHER) tok = pair_token[bm * 128 + arow];
    else        tok = bm * 128 + arow;
    const bf16* asrc = X + (size_t)tok * DDIM + ahalf * 16;

    const int brow  = tid >> 3;                        // k within chunk, 0..31
    const int bcol0 = (tid & 7) * 8;                   // n within tile

    const unsigned offG[2] = {(unsigned)(uintptr_t)&stG[0][0],
                              (unsigned)(uintptr_t)&stG[1][0]};
    const unsigned offU[2] = {(unsigned)(uintptr_t)&stU[0][0],
                              (unsigned)(uintptr_t)&stU[1][0]};

    // prologue: TDM fetch of K-chunk 0 (wave 0 issues; TDM ignores EXEC)
    if (wave == 0) {
        tdm_load_tile_f32(offG[0], G + n0, (unsigned)Ntot, DDIM, 64, 32);
        tdm_load_tile_f32(offU[0], U + n0, (unsigned)Ntot, DDIM, 64, 32);
    }

    v8f accG[2][2] = {};
    v8f accU[2][2] = {};

    int pp = 0;
    for (int kc = 0; kc < DDIM; kc += 32, pp ^= 1) {
        const bool pre = (kc + 32 < DDIM);
        if (wave == 0 && pre) {                        // prefetch next chunk
            tdm_load_tile_f32(offG[pp ^ 1], G + (size_t)(kc + 32) * Ntot + n0,
                              (unsigned)Ntot, DDIM, 64, 32);
            tdm_load_tile_f32(offU[pp ^ 1], U + (size_t)(kc + 32) * Ntot + n0,
                              (unsigned)Ntot, DDIM, 64, 32);
        }

        // A chunk: global -> regs (gathered rows)
        const uint4* s = (const uint4*)(asrc + kc);
        const uint4 a0 = s[0];
        const uint4 a1 = s[1];
        __builtin_prefetch(asrc + kc + 32, 0, 1);

        // wait for current chunk's 2 TDM ops (2 newer in flight when prefetching)
        if (pre) __builtin_amdgcn_s_wait_tensorcnt(2);
        else     __builtin_amdgcn_s_wait_tensorcnt(0);
        __syncthreads();                               // stage[pp] visible; lA/lB free

        *(uint4*)(&lA[arow * 32 + ahalf * 16])     = a0;
        *(uint4*)(&lA[arow * 32 + ahalf * 16 + 8]) = a1;

        const float* gs = &stG[pp][brow * 64 + bcol0];
        const float* us = &stU[pp][brow * 64 + bcol0];
        #pragma unroll
        for (int j = 0; j < 8; ++j) {
            lBg[(bcol0 + j) * 32 + brow] = (bf16)gs[j];
            lBu[(bcol0 + j) * 32 + brow] = (bf16)us[j];
        }
        __syncthreads();

        v16bf af[2], gf[2], uf[2];
        af[0] = frag_from_lds(lA,  wm * 32,      lane);
        af[1] = frag_from_lds(lA,  wm * 32 + 16, lane);
        gf[0] = frag_from_lds(lBg, wn * 32,      lane);
        gf[1] = frag_from_lds(lBg, wn * 32 + 16, lane);
        uf[0] = frag_from_lds(lBu, wn * 32,      lane);
        uf[1] = frag_from_lds(lBu, wn * 32 + 16, lane);

        #pragma unroll
        for (int i = 0; i < 2; ++i)
            #pragma unroll
            for (int j = 0; j < 2; ++j) {
                accG[i][j] = WMMA_BF16(af[i], gf[j], accG[i][j]);
                accU[i][j] = WMMA_BF16(af[i], uf[j], accU[i][j]);
            }
    }

    // epilogue: h = silu(g) * u  -> bf16
    const int mrow0 = bm * 128 + wm * 32;
    const int lm    = ((lane >> 4) << 3);
    const int ln    = lane & 15;
    #pragma unroll
    for (int i = 0; i < 2; ++i)
        #pragma unroll
        for (int j = 0; j < 2; ++j)
            #pragma unroll
            for (int v = 0; v < 8; ++v) {
                const int m = mrow0 + i * 16 + v + lm;
                const int n = n0 + wn * 32 + j * 16 + ln;
                const float h = silu_fast(accG[i][j][v]) * accU[i][j][v];
                Hout[(size_t)m * Ntot + n] = (bf16)h;
            }
}

// ----------------------------- down GEMM ------------------------------------
// ROUTED=true : out[token] += w_pair * (h_pair @ down_e)  (atomic f32)
// ROUTED=false: out[token]  = s_buf @ sh_down             (plain store; runs first)

template <bool ROUTED>
__global__ __launch_bounds__(256)
void down_kernel(const bf16* __restrict__ Hbuf,
                 const float* __restrict__ Wall,
                 const int* __restrict__ pair_token,
                 const float* __restrict__ pair_weight,
                 const int* __restrict__ tile_expert,
                 float* __restrict__ Out,
                 int Kd) {
    __shared__ __align__(16) float stW[2][32 * 64];   // TDM landing zone (fp32)
    __shared__ __align__(16) bf16  lA[128 * 32];
    __shared__ __align__(16) bf16  lB[64 * 32];       // transposed: [n][k]

    const int bn = blockIdx.x;
    const int bm = blockIdx.y;
    const int n0 = bn * 64;

    const float* W;
    if (ROUTED) {
        const int e = tile_expert[bm];
        if (e < 0) return;
        W = Wall + (size_t)e * Kd * DDIM;
    } else {
        W = Wall;
    }

    const int tid  = threadIdx.x;
    const int lane = tid & 31;
    const int wave = tid >> 5;
    const int wm   = wave >> 1;
    const int wn   = wave & 1;

    const int arow  = tid >> 1;
    const int ahalf = tid & 1;
    const bf16* asrc = Hbuf + (size_t)(bm * 128 + arow) * Kd + ahalf * 16;

    const int brow  = tid >> 3;
    const int bcol0 = (tid & 7) * 8;

    const unsigned offW[2] = {(unsigned)(uintptr_t)&stW[0][0],
                              (unsigned)(uintptr_t)&stW[1][0]};

    if (wave == 0)
        tdm_load_tile_f32(offW[0], W + n0, DDIM, (unsigned)Kd, 64, 32);

    v8f acc[2][2] = {};

    int pp = 0;
    for (int kc = 0; kc < Kd; kc += 32, pp ^= 1) {
        const bool pre = (kc + 32 < Kd);
        if (wave == 0 && pre)
            tdm_load_tile_f32(offW[pp ^ 1], W + (size_t)(kc + 32) * DDIM + n0,
                              DDIM, (unsigned)Kd, 64, 32);

        const uint4* s = (const uint4*)(asrc + kc);
        const uint4 a0 = s[0];
        const uint4 a1 = s[1];
        __builtin_prefetch(asrc + kc + 32, 0, 1);

        if (pre) __builtin_amdgcn_s_wait_tensorcnt(1);
        else     __builtin_amdgcn_s_wait_tensorcnt(0);
        __syncthreads();

        *(uint4*)(&lA[arow * 32 + ahalf * 16])     = a0;
        *(uint4*)(&lA[arow * 32 + ahalf * 16 + 8]) = a1;

        const float* wsrc = &stW[pp][brow * 64 + bcol0];
        #pragma unroll
        for (int j = 0; j < 8; ++j)
            lB[(bcol0 + j) * 32 + brow] = (bf16)wsrc[j];
        __syncthreads();

        v16bf af[2], bfg[2];
        af[0]  = frag_from_lds(lA, wm * 32,      lane);
        af[1]  = frag_from_lds(lA, wm * 32 + 16, lane);
        bfg[0] = frag_from_lds(lB, wn * 32,      lane);
        bfg[1] = frag_from_lds(lB, wn * 32 + 16, lane);

        #pragma unroll
        for (int i = 0; i < 2; ++i)
            #pragma unroll
            for (int j = 0; j < 2; ++j)
                acc[i][j] = WMMA_BF16(af[i], bfg[j], acc[i][j]);
    }

    const int mrow0 = bm * 128 + wm * 32;
    const int lm    = ((lane >> 4) << 3);
    const int ln    = lane & 15;
    #pragma unroll
    for (int i = 0; i < 2; ++i)
        #pragma unroll
        for (int j = 0; j < 2; ++j)
            #pragma unroll
            for (int v = 0; v < 8; ++v) {
                const int m = mrow0 + i * 16 + v + lm;
                const int n = n0 + wn * 32 + j * 16 + ln;
                const float val = acc[i][j][v];
                if (ROUTED) {
                    const float w = pair_weight[m];          // padding rows: w == 0
                    const int   t = pair_token[m];
                    atomicAdd(&Out[(size_t)t * DDIM + n], val * w);
                } else {
                    Out[(size_t)m * DDIM + n] = val;
                }
            }
}

// ----------------------------- launcher -------------------------------------

extern "C" void kernel_launch(void* const* d_in, const int* in_sizes, int n_in,
                              void* d_out, int out_size, void* d_ws, size_t ws_size,
                              hipStream_t stream) {
    const float* x      = (const float*)d_in[0];   // [1,4096,1024]
    const float* gate_w = (const float*)d_in[1];   // [1024,64]
    const float* gate_b = (const float*)d_in[2];   // [64]
    const float* gproj  = (const float*)d_in[3];   // [64,1024,768]
    const float* uproj  = (const float*)d_in[4];   // [64,1024,768]
    const float* dproj  = (const float*)d_in[5];   // [64,768,1024]
    const float* shg    = (const float*)d_in[6];   // [1024,1536]
    const float* shu    = (const float*)d_in[7];   // [1024,1536]
    const float* shd    = (const float*)d_in[8];   // [1536,1024]
    float* out = (float*)d_out;                    // [1,4096,1024] f32

    char* w = (char*)d_ws;
    auto alloc = [&](size_t bytes) -> char* {
        char* p = w;
        w += (bytes + 255) & ~(size_t)255;
        return p;
    };
    int*   topk_idx     = (int*)  alloc((size_t)T_TOK * TOPK * 4);
    float* topk_w       = (float*)alloc((size_t)T_TOK * TOPK * 4);
    int*   expert_count = (int*)  alloc(NEXP * 4);
    int*   pad_off      = (int*)  alloc((NEXP + 1) * 4);
    int*   cursor       = (int*)  alloc(NEXP * 4);
    int*   tile_expert  = (int*)  alloc(MTILES * 4);
    int*   pair_token   = (int*)  alloc((size_t)PMAX * 4);
    float* pair_weight  = (float*)alloc((size_t)PMAX * 4);
    bf16*  x_bf         = (bf16*) alloc((size_t)T_TOK * DDIM * 2);
    bf16*  h_buf        = (bf16*) alloc((size_t)PMAX * HDIM * 2);
    bf16*  s_buf        = (bf16*) alloc((size_t)T_TOK * SHDIM * 2);

    // 1) routing
    init_kernel<<<1, 64, 0, stream>>>(expert_count);
    gate_topk_kernel<<<T_TOK, 64, 0, stream>>>(x, gate_w, gate_b, topk_idx, topk_w, expert_count);
    scan_kernel<<<1, 256, 0, stream>>>(expert_count, pad_off, cursor, tile_expert,
                                       pair_token, pair_weight);
    scatter_kernel<<<(T_TOK * TOPK) / 256, 256, 0, stream>>>(topk_idx, topk_w, cursor,
                                                             pair_token, pair_weight);
    // 2) activations to bf16
    xcvt_kernel<<<(T_TOK * DDIM) / (256 * 4), 256, 0, stream>>>(x, x_bf);

    // 3) routed experts: h = silu(x@g) * (x@u), gathered per expert tile
    gateup_kernel<true><<<dim3(HDIM / 64, MTILES), 256, 0, stream>>>(
        x_bf, gproj, uproj, pair_token, tile_expert, h_buf, HDIM);

    // 4) shared expert: s = silu(x@shg) * (x@shu)
    gateup_kernel<false><<<dim3(SHDIM / 64, T_TOK / 128), 256, 0, stream>>>(
        x_bf, shg, shu, nullptr, nullptr, s_buf, SHDIM);

    // 5) shared down-proj writes the output (no zero-init needed)
    down_kernel<false><<<dim3(DDIM / 64, T_TOK / 128), 256, 0, stream>>>(
        s_buf, shd, nullptr, nullptr, nullptr, out, SHDIM);

    // 6) routed down-proj accumulates on top with weight scaling (f32 atomics)
    down_kernel<true><<<dim3(DDIM / 64, MTILES), 256, 0, stream>>>(
        h_buf, dproj, pair_token, pair_weight, tile_expert, out, HDIM);
}
